// MultiheadAttention_85839216378125
// MI455X (gfx1250) — matrix operand order repaired and checked by reading, not compiled, not run on hardware
//
#include <hip/hip_runtime.h>
#include <hip/hip_bf16.h>

// ---------------------------------------------------------------------------
// Multihead attention, B=1, S=4096, QW=HID=512, HEADS=8, D=64.
// All GEMMs + attention matmuls run on v_wmma_f32_16x16x32_bf16 (wave32).
// Round 2: DPP ROW_XMASK reductions (no ds_bpermute/s_wait_dscnt), and
// __launch_bounds__(32,1) on the attention kernel to kill scratch spills.
// ---------------------------------------------------------------------------

#define S_LEN   4096
#define QW_DIM  512
#define HID_DIM 512
#define N_HEADS 8
#define HEAD_D  64
#define ATTN_SCALE 0.125f   // 1/sqrt(64)

typedef __attribute__((ext_vector_type(16))) __bf16 v16bf;
typedef __attribute__((ext_vector_type(8)))  float  v8f;

union AFrag {
    unsigned int u[8];
    v16bf v;
};

__device__ __forceinline__ unsigned short f32_to_bf16(float f) {
    unsigned int u = __float_as_uint(f);
    unsigned int r = u + 0x7FFFu + ((u >> 16) & 1u);   // round-to-nearest-even
    return (unsigned short)(r >> 16);
}

// ---- DPP ROW_XMASK lane-xor within 16-lane rows: pure VALU, no LDS ----
template <int XMASK>
__device__ __forceinline__ float dpp_xor16(float x) {
    return __int_as_float(__builtin_amdgcn_update_dpp(
        0, __float_as_int(x), 0x160 | XMASK, 0xf, 0xf, true));
}
__device__ __forceinline__ float redmax16(float x) {
    x = fmaxf(x, dpp_xor16<1>(x));
    x = fmaxf(x, dpp_xor16<2>(x));
    x = fmaxf(x, dpp_xor16<4>(x));
    x = fmaxf(x, dpp_xor16<8>(x));
    return x;
}
__device__ __forceinline__ float redsum16(float x) {
    x += dpp_xor16<1>(x);
    x += dpp_xor16<2>(x);
    x += dpp_xor16<4>(x);
    x += dpp_xor16<8>(x);
    return x;
}

// --------------------------- f32 -> bf16 convert ---------------------------
__global__ void cvt_f32_bf16_kernel(const float* __restrict__ in,
                                    unsigned short* __restrict__ out, int n) {
    int i = blockIdx.x * blockDim.x + threadIdx.x;
    if (i < n) out[i] = f32_to_bf16(in[i]);
}

// ------------------------------- WMMA GEMM ---------------------------------
// C[m,n] = sum_k A[m,k] * Bw[n,k]   (A: MxK bf16 row-major, Bw: NxK bf16 row-major)
// One wave per 16x16 output tile. MODE 0: bf16 row-major store.
// MODE 1: bf16 transposed store (out[n*ldc + m]).  MODE 2: f32 row-major store.
template <int MODE>
__global__ __launch_bounds__(256) void
wmma_gemm_kernel(const unsigned short* __restrict__ A,
                 const unsigned short* __restrict__ Bw,
                 void* __restrict__ Cout,
                 int Kdim, int lda, int ldb, int ldc) {
    const int lane = threadIdx.x & 31;
    const int wave = threadIdx.x >> 5;
    const int half = lane >> 4;
    const int ln   = lane & 15;
    const int mt   = blockIdx.x;               // M tile
    const int nt   = blockIdx.y * 8 + wave;    // N tile (8 waves per block)

    const unsigned short* arowp = A  + (size_t)(mt * 16 + ln) * lda;
    const unsigned short* browp = Bw + (size_t)(nt * 16 + ln) * ldb;

    v8f acc = {};
    for (int k0 = 0; k0 < Kdim; k0 += 32) {
        AFrag fa, fb;
#pragma unroll
        for (int v = 0; v < 8; ++v) {
            int ka = k0 + (v < 4 ? 2 * v : 2 * v + 8) + 8 * half;   // A layout
            int kb = k0 + (v < 4 ? 2 * v : 2 * v + 8) + 8 * half;   // B layout
            fa.u[v] = *(const unsigned int*)(arowp + ka);
            fb.u[v] = *(const unsigned int*)(browp + kb);
        }
        acc = __builtin_amdgcn_wmma_f32_16x16x32_bf16(
            false, fa.v, false, fb.v, (short)0, acc, false, false);
    }

    // D layout: VGPR r -> M = r + 8*half, N = ln
    if (MODE == 2) {
        float* Cf = (float*)Cout;
#pragma unroll
        for (int r = 0; r < 8; ++r)
            Cf[(size_t)(mt * 16 + r + 8 * half) * ldc + nt * 16 + ln] = acc[r];
    } else if (MODE == 1) {
        unsigned short* Cb = (unsigned short*)Cout;
#pragma unroll
        for (int r = 0; r < 8; ++r)
            Cb[(size_t)(nt * 16 + ln) * ldc + mt * 16 + r + 8 * half] =
                f32_to_bf16(acc[r]);
    } else {
        unsigned short* Cb = (unsigned short*)Cout;
#pragma unroll
        for (int r = 0; r < 8; ++r)
            Cb[(size_t)(mt * 16 + r + 8 * half) * ldc + nt * 16 + ln] =
                f32_to_bf16(acc[r]);
    }
}

// --------------------------- flash attention -------------------------------
// grid = (S/16 query tiles, HEADS), block = 32 (one wave).
// Q,K: bf16 (S, HID) row-major. Vt: bf16 (HID, S) row-major (transposed).
// ctx out: bf16 (S, HID) row-major.
__global__ __launch_bounds__(32, 1) void
flash_attn_kernel(const unsigned short* __restrict__ Q,
                  const unsigned short* __restrict__ K,
                  const unsigned short* __restrict__ Vt,
                  unsigned short* __restrict__ ctx) {
    __shared__ unsigned short p_lds[16 * 32];   // P tile staging (1 KB)

    const int lane = threadIdx.x & 31;
    const int half = lane >> 4;
    const int ln   = lane & 15;
    const int qt   = blockIdx.x;
    const int head = blockIdx.y;
    const int hoff = head * HEAD_D;

    // Load the 16x64 Q tile as two A fragments (K-steps of 32)
    v16bf qa[2];
#pragma unroll
    for (int kk = 0; kk < 2; ++kk) {
        AFrag fa;
#pragma unroll
        for (int v = 0; v < 8; ++v) {
            int k = 32 * kk + (v < 4 ? 2 * v : 2 * v + 8) + 8 * half;
            fa.u[v] = *(const unsigned int*)(Q + (size_t)(qt * 16 + ln) * HID_DIM + hoff + k);
        }
        qa[kk] = fa.v;
    }

    v8f   Oacc[4] = {};
    float mrow[8], lrow[8];
#pragma unroll
    for (int r = 0; r < 8; ++r) { mrow[r] = -1e30f; lrow[r] = 0.0f; }

    const unsigned short* Kbase = K + hoff;               // + key*HID
    const unsigned short* Vbase = Vt + (size_t)hoff * S_LEN;

    for (int it = 0; it < S_LEN / 32; ++it) {
        const int kbase = it * 32;

        // ---- S tile (16 queries x 32 keys) = Q (16x64) x K^T (64x32) ----
        v8f cc[2] = {};
#pragma unroll
        for (int t = 0; t < 2; ++t) {
            const unsigned short* krow = Kbase + (size_t)(kbase + t * 16 + ln) * HID_DIM;
#pragma unroll
            for (int kk = 0; kk < 2; ++kk) {
                AFrag fb;
#pragma unroll
                for (int v = 0; v < 8; ++v) {
                    int d = 32 * kk + (v < 4 ? 2 * v : 2 * v + 8) + 8 * half;
                    fb.u[v] = *(const unsigned int*)(krow + d);
                }
                cc[t] = __builtin_amdgcn_wmma_f32_16x16x32_bf16(
                    false, qa[kk], false, fb.v, (short)0, cc[t], false, false);
            }
        }

        // ---- online softmax over this 32-key chunk (DPP reductions) ----
#pragma unroll
        for (int r = 0; r < 8; ++r) {
            float s0 = cc[0][r] * ATTN_SCALE;
            float s1 = cc[1][r] * ATTN_SCALE;
            float rm = redmax16(fmaxf(s0, s1));
            float mnew  = fmaxf(mrow[r], rm);
            float alpha = __expf(mrow[r] - mnew);
            float p0 = __expf(s0 - mnew);
            float p1 = __expf(s1 - mnew);
            float rs = redsum16(p0 + p1);
            lrow[r] = lrow[r] * alpha + rs;
            mrow[r] = mnew;
#pragma unroll
            for (int j = 0; j < 4; ++j) Oacc[j][r] *= alpha;
            int row = r + 8 * half;
            p_lds[row * 32 + ln]      = f32_to_bf16(p0);
            p_lds[row * 32 + 16 + ln] = f32_to_bf16(p1);
        }
        __syncthreads();

        // reload P (16x32) as an A fragment
        AFrag pa;
#pragma unroll
        for (int v = 0; v < 8; ++v) {
            int k = (v < 4 ? 2 * v : 2 * v + 8) + 8 * half;
            pa.u[v] = *(const unsigned int*)(p_lds + ln * 32 + k);
        }
        __syncthreads();

        // ---- O (16x64) += P (16x32) x V (32x64), via Vt contiguous rows ----
#pragma unroll
        for (int j = 0; j < 4; ++j) {
            const unsigned short* vrow = Vbase + (size_t)(j * 16 + ln) * S_LEN + kbase;
            AFrag fv;
#pragma unroll
            for (int v = 0; v < 8; ++v) {
                int kv = (v < 4 ? 2 * v : 2 * v + 8) + 8 * half;
                fv.u[v] = *(const unsigned int*)(vrow + kv);
            }
            Oacc[j] = __builtin_amdgcn_wmma_f32_16x16x32_bf16(
                false, pa.v, false, fv.v, (short)0, Oacc[j], false, false);
        }
    }

    // ---- normalize and store ctx (bf16, row-major S x HID) ----
#pragma unroll
    for (int r = 0; r < 8; ++r) {
        float inv = 1.0f / lrow[r];
        int row = qt * 16 + r + 8 * half;
#pragma unroll
        for (int j = 0; j < 4; ++j)
            ctx[(size_t)row * HID_DIM + hoff + j * 16 + ln] =
                f32_to_bf16(Oacc[j][r] * inv);
    }
}

// ------------------------------ launch glue --------------------------------
extern "C" void kernel_launch(void* const* d_in, const int* in_sizes, int n_in,
                              void* d_out, int out_size, void* d_ws, size_t ws_size,
                              hipStream_t stream) {
    const float* xq      = (const float*)d_in[0];   // (1, 4096, 512)
    const float* xk      = (const float*)d_in[1];
    const float* xv      = (const float*)d_in[2];
    const float* in_proj = (const float*)d_in[3];   // (1536, 512)
    const float* w_out   = (const float*)d_in[4];   // (512, 512)

    const size_t X_ELTS  = (size_t)S_LEN * QW_DIM;       // 2M
    const size_t WP_ELTS = (size_t)3 * HID_DIM * QW_DIM; // 768K
    const size_t WO_ELTS = (size_t)HID_DIM * HID_DIM;    // 256K

    unsigned short* ws = (unsigned short*)d_ws;     // bf16 workspace regions
    unsigned short* Xq   = ws;                 ws += X_ELTS;
    unsigned short* Xk   = ws;                 ws += X_ELTS;
    unsigned short* Xv   = ws;                 ws += X_ELTS;
    unsigned short* Wp   = ws;                 ws += WP_ELTS;
    unsigned short* Wo   = ws;                 ws += WO_ELTS;
    unsigned short* Qb   = ws;                 ws += X_ELTS;
    unsigned short* Kb   = ws;                 ws += X_ELTS;
    unsigned short* Vtb  = ws;                 ws += X_ELTS;   // transposed (HID, S)
    unsigned short* Ctx  = ws;                 ws += X_ELTS;

    // 1) convert everything to bf16
    cvt_f32_bf16_kernel<<<(X_ELTS  + 255) / 256, 256, 0, stream>>>(xq, Xq, (int)X_ELTS);
    cvt_f32_bf16_kernel<<<(X_ELTS  + 255) / 256, 256, 0, stream>>>(xk, Xk, (int)X_ELTS);
    cvt_f32_bf16_kernel<<<(X_ELTS  + 255) / 256, 256, 0, stream>>>(xv, Xv, (int)X_ELTS);
    cvt_f32_bf16_kernel<<<(WP_ELTS + 255) / 256, 256, 0, stream>>>(in_proj, Wp, (int)WP_ELTS);
    cvt_f32_bf16_kernel<<<(WO_ELTS + 255) / 256, 256, 0, stream>>>(w_out, Wo, (int)WO_ELTS);

    // 2) QKV projections: 4096x512 @ (512x512)^T
    dim3 ggrid(S_LEN / 16, HID_DIM / 16 / 8);   // (256, 4), 8 waves/block
    dim3 gblock(256);
    wmma_gemm_kernel<0><<<ggrid, gblock, 0, stream>>>(
        Xq, Wp,                                Qb,  QW_DIM, QW_DIM, QW_DIM, HID_DIM);
    wmma_gemm_kernel<0><<<ggrid, gblock, 0, stream>>>(
        Xk, Wp + (size_t)HID_DIM * QW_DIM,     Kb,  QW_DIM, QW_DIM, QW_DIM, HID_DIM);
    wmma_gemm_kernel<1><<<ggrid, gblock, 0, stream>>>(
        Xv, Wp + (size_t)2 * HID_DIM * QW_DIM, Vtb, QW_DIM, QW_DIM, QW_DIM, S_LEN);

    // 3) flash attention: one wave per (16-query tile, head)
    dim3 agrid(S_LEN / 16, N_HEADS);
    flash_attn_kernel<<<agrid, 32, 0, stream>>>(Qb, Kb, Vtb, Ctx);

    // 4) output projection: ctx @ w_out^T -> f32 out
    wmma_gemm_kernel<2><<<ggrid, gblock, 0, stream>>>(
        Ctx, Wo, d_out, HID_DIM, HID_DIM, HID_DIM, HID_DIM);

    (void)in_sizes; (void)n_in; (void)out_size; (void)ws_size;
}
